// OursBlockWrapper_6700148982396
// MI455X (gfx1250) — matrix operand order repaired and checked by reading, not compiled, not run on hardware
//
#include <hip/hip_runtime.h>
#include <hip/hip_bf16.h>
#include <math.h>

// ---------------------------------------------------------------------------
// Problem constants (from the reference)
// ---------------------------------------------------------------------------
#define BB   64
#define TT   197
#define CC   1024
#define NH   16
#define HDIM 64
#define HIDN 4096
#define KK   157            // T - requested_r = 197 - 40
#define BT   (BB*TT)        // 12608
#define BK   (BB*KK)        // 10048

typedef __attribute__((ext_vector_type(16))) __bf16 v16bf;
typedef __attribute__((ext_vector_type(8)))  float  v8f;

__device__ __forceinline__ __bf16 f2bf(float f) {
  union { float f; unsigned u; } c; c.f = f;
  unsigned r = c.u + 0x7fffu + ((c.u >> 16) & 1u);   // round-to-nearest-even
  union { unsigned short s; __bf16 b; } o; o.s = (unsigned short)(r >> 16);
  return o.b;
}

// Async global->LDS copy, 16B per lane, tracked by ASYNCcnt (CDNA5).
__device__ __forceinline__ void async_b128(unsigned lds_byte_addr, const void* gaddr) {
  asm volatile("global_load_async_to_lds_b128 %0, %1, off"
               :: "v"(lds_byte_addr), "v"(gaddr) : "memory");
}

// ---------------------------------------------------------------------------
// fp32 -> bf16 bulk conversion (weights), 4 elements per thread
// ---------------------------------------------------------------------------
__global__ void k_w2bf(const float* __restrict__ w, __bf16* __restrict__ o, int n) {
  int i = (blockIdx.x * 256 + threadIdx.x) * 4;
  if (i < n) {
    float4 f = *(const float4*)(w + i);
    o[i] = f2bf(f.x); o[i + 1] = f2bf(f.y); o[i + 2] = f2bf(f.z); o[i + 3] = f2bf(f.w);
  }
}

// ---------------------------------------------------------------------------
// LayerNorm (fp32 in -> bf16 out), one block per row of length CC
// ---------------------------------------------------------------------------
__global__ void k_layernorm_bf16(const float* __restrict__ in,
                                 const float* __restrict__ g,
                                 const float* __restrict__ bta,
                                 __bf16* __restrict__ out) {
  int row = blockIdx.x;
  const float* x = in + (size_t)row * CC;
  __shared__ float red[256];
  float s = 0.f;
  for (int c = threadIdx.x; c < CC; c += 256) s += x[c];
  red[threadIdx.x] = s; __syncthreads();
  for (int o = 128; o > 0; o >>= 1) {
    if (threadIdx.x < o) red[threadIdx.x] += red[threadIdx.x + o];
    __syncthreads();
  }
  float mu = red[0] * (1.f / CC);
  __syncthreads();
  float v = 0.f;
  for (int c = threadIdx.x; c < CC; c += 256) { float d = x[c] - mu; v += d * d; }
  red[threadIdx.x] = v; __syncthreads();
  for (int o = 128; o > 0; o >>= 1) {
    if (threadIdx.x < o) red[threadIdx.x] += red[threadIdx.x + o];
    __syncthreads();
  }
  float inv = rsqrtf(red[0] * (1.f / CC) + 1e-6f);
  for (int c = threadIdx.x; c < CC; c += 256)
    out[(size_t)row * CC + c] = f2bf((x[c] - mu) * inv * g[c] + bta[c]);
}

// ---------------------------------------------------------------------------
// WMMA GEMM with async-LDS double buffering:
//   out[M,N] = A_bf16[M,K] @ B_bf16[N,K]^T (+bias)(+GELU)(+residual)
// Block: 256 threads = 8 waves; tile M=128, N=64, K-step 32.
// Tiles staged with GLOBAL_LOAD_ASYNC_TO_LDS_B128 (3 chunks/thread/tile),
// ping-pong buffers, s_wait_asynccnt + split barriers.
// Row pitch 72 bf16 = 144 B: 16B-aligned for B128, conflict-free frag reads.
// ---------------------------------------------------------------------------
#define GP 72
__global__ void k_gemm_wmma(const __bf16* __restrict__ A,
                            const __bf16* __restrict__ Bw,
                            const float* __restrict__ bias,
                            const float* __restrict__ resid,
                            float* __restrict__ outf,
                            __bf16* __restrict__ outb,
                            int M, int N, int Kd, int act_gelu) {
  __shared__ __bf16 sT[2][(128 + 64) * GP];     // A rows 0..127, B rows 128..191
  int tid = threadIdx.x;
  int wave = tid >> 5, lane = tid & 31;
  int half = lane >> 4, l16 = lane & 15;
  int tm = blockIdx.y * 128;
  int tn = blockIdx.x * 64;

  // Per-thread staging assignment: 768 16-byte chunks per 32-wide K tile
  // (A: 128 rows x 4 chunks, B: 64 rows x 4 chunks) -> 3 chunks/thread.
  const __bf16* gsrc[3];
  unsigned loff[3];
#pragma unroll
  for (int i = 0; i < 3; ++i) {
    int idx = tid + i * 256;
    if (idx < 512) {                            // A chunk
      int r = idx >> 2, c = idx & 3;
      int gm = tm + r; if (gm >= M) gm = M - 1; // clamp: rows >= M never stored
      gsrc[i] = A + (size_t)gm * Kd + c * 8;
      loff[i] = (unsigned)(r * (GP * 2) + c * 16);
    } else {                                    // B chunk
      int j = idx - 512;
      int r = j >> 2, c = j & 3;
      int gn = tn + r; if (gn >= N) gn = N - 1;
      gsrc[i] = Bw + (size_t)gn * Kd + c * 8;
      loff[i] = (unsigned)((128 + r) * (GP * 2) + c * 16);
    }
  }
  unsigned base0 = (unsigned)(size_t)&sT[0][0];
  unsigned base1 = (unsigned)(size_t)&sT[1][0];

  v8f acc[4];
  for (int i = 0; i < 4; ++i) for (int j = 0; j < 8; ++j) acc[i][j] = 0.f;

  // prologue: stage tile 0 into buffer 0
#pragma unroll
  for (int i = 0; i < 3; ++i) async_b128(base0 + loff[i], gsrc[i]);

  int nk = Kd >> 5;
  for (int kt = 0; kt < nk; ++kt) {
    unsigned curb = (kt & 1) ? base1 : base0;
    if (kt + 1 < nk) {
      unsigned nxtb = (kt & 1) ? base0 : base1;
#pragma unroll
      for (int i = 0; i < 3; ++i)
        async_b128(nxtb + loff[i], (const char*)gsrc[i] + (size_t)(kt + 1) * 64);
      asm volatile("s_wait_asynccnt 3" ::: "memory");  // tile kt landed (in-order)
    } else {
      asm volatile("s_wait_asynccnt 0" ::: "memory");
    }
    __syncthreads();                                   // all waves' chunks visible

    const __bf16* bufA = &sT[kt & 1][0];
    const __bf16* bufB = &sT[kt & 1][128 * GP];

    v16bf af;
    int mrow = wave * 16 + l16;
#pragma unroll
    for (int v = 0; v < 8; ++v) {
      int kb = (v & 3) * 2 + (v >> 2) * 16 + half * 8;
      af[2 * v]     = bufA[mrow * GP + kb];
      af[2 * v + 1] = bufA[mrow * GP + kb + 1];
    }
#pragma unroll
    for (int nt = 0; nt < 4; ++nt) {
      v16bf bfr;
#pragma unroll
      for (int v = 0; v < 8; ++v) {
        int kc = (v & 3) * 2 + (v >> 2) * 16 + half * 8;
        bfr[2 * v]     = bufB[(nt * 16 + l16) * GP + kc];
        bfr[2 * v + 1] = bufB[(nt * 16 + l16) * GP + kc + 1];
      }
      acc[nt] = __builtin_amdgcn_wmma_f32_16x16x32_bf16(
          false, af, false, bfr, (short)0, acc[nt], false, false);
    }
    __syncthreads();                                   // done reading cur buffer
  }

#pragma unroll
  for (int nt = 0; nt < 4; ++nt) {
    int n = tn + nt * 16 + l16;
#pragma unroll
    for (int v = 0; v < 8; ++v) {
      int m = tm + wave * 16 + v + half * 8;
      if (m < M && n < N) {
        float val = acc[nt][v];
        if (bias) val += bias[n];
        if (act_gelu) val = 0.5f * val * (1.f + erff(val * 0.70710678118654752f));
        if (resid) val += resid[(size_t)m * N + n];
        if (outf) outf[(size_t)m * N + n] = val;
        if (outb) outb[(size_t)m * N + n] = f2bf(val);
      }
    }
  }
}

// ---------------------------------------------------------------------------
// Attention: one block per (b,h). qkv bf16 [B*T, 3C].  o bf16 [B*T, C].
// Per wave: 16 queries; QK^T via WMMA (keys padded to 224), shuffle softmax,
// P in wave-private LDS, then P@V via WMMA. Index clamping (no branches):
// padded keys are masked to -inf (QK^T) or weighted by exactly 0 (P@V).
// ---------------------------------------------------------------------------
__global__ void k_attention(const __bf16* __restrict__ qkv,
                            __bf16* __restrict__ o) {
  const int LDP = 224;
  __shared__ __bf16 sP[8 * 16 * 224];           // 56 KB, per-wave strips
  int b = blockIdx.x / NH, h = blockIdx.x % NH;
  int tid = threadIdx.x, wave = tid >> 5, lane = tid & 31;
  int half = lane >> 4, l16 = lane & 15;
  __bf16* P = sP + wave * 16 * LDP;
  const size_t row3 = (size_t)3 * CC;

  for (int qt = wave; qt < 13; qt += 8) {       // ceil(197/16) = 13 q-tiles
    // ---- Q fragments (A matrix 16x64, two K-steps of 32) ----
    v16bf qf[2];
    int t = qt * 16 + l16; if (t >= TT) t = TT - 1;         // clamp
    size_t qbase = (size_t)(b * TT + t) * row3 + h * HDIM;
#pragma unroll
    for (int ks = 0; ks < 2; ++ks)
#pragma unroll
      for (int v = 0; v < 8; ++v) {
        int kb = (v & 3) * 2 + (v >> 2) * 16 + half * 8 + ks * 32;
        qf[ks][2 * v]     = qkv[qbase + kb];
        qf[ks][2 * v + 1] = qkv[qbase + kb + 1];
      }

    // ---- logits = Q @ K^T over 14 key tiles ----
    v8f lacc[14];
    for (int nt = 0; nt < 14; ++nt) for (int j = 0; j < 8; ++j) lacc[nt][j] = 0.f;
#pragma unroll
    for (int nt = 0; nt < 14; ++nt) {
      int key = nt * 16 + l16; if (key >= TT) key = TT - 1; // clamp; col masked later
      size_t kb0 = (size_t)(b * TT + key) * row3 + CC + h * HDIM;
#pragma unroll
      for (int ks = 0; ks < 2; ++ks) {
        v16bf kf;
#pragma unroll
        for (int v = 0; v < 8; ++v) {
          int dd = ks * 32 + (v & 3) * 2 + (v >> 2) * 16 + half * 8;
          kf[2 * v]     = qkv[kb0 + dd];
          kf[2 * v + 1] = qkv[kb0 + dd + 1];
        }
        lacc[nt] = __builtin_amdgcn_wmma_f32_16x16x32_bf16(
            false, qf[ks], false, kf, (short)0, lacc[nt], false, false);
      }
    }

    // ---- softmax over 224 keys (pad masked); rows live in lane halves ----
#pragma unroll
    for (int v = 0; v < 8; ++v) {
      float mx = -3.0e38f;
#pragma unroll
      for (int nt = 0; nt < 14; ++nt) {
        int n = nt * 16 + l16;
        float val = lacc[nt][v] * 0.125f;       // 1/sqrt(64)
        if (n >= TT) val = -3.0e38f;
        lacc[nt][v] = val;
        mx = fmaxf(mx, val);
      }
#pragma unroll
      for (int off = 8; off >= 1; off >>= 1)
        mx = fmaxf(mx, __shfl_xor(mx, off, 32));
      float sum = 0.f;
#pragma unroll
      for (int nt = 0; nt < 14; ++nt) {
        int n = nt * 16 + l16;
        float e = (n < TT) ? __expf(lacc[nt][v] - mx) : 0.f;
        lacc[nt][v] = e; sum += e;
      }
#pragma unroll
      for (int off = 8; off >= 1; off >>= 1)
        sum += __shfl_xor(sum, off, 32);
      float inv = 1.f / sum;
      int m = v + half * 8;
#pragma unroll
      for (int nt = 0; nt < 14; ++nt)
        P[m * LDP + nt * 16 + l16] = f2bf(lacc[nt][v] * inv);
    }
    // same-wave LDS RAW: LDS is in-order per wave; make the wait explicit
    asm volatile("s_wait_dscnt 0" ::: "memory");

    // ---- O = P @ V  (contract over 224 keys, 7 K-steps) ----
    v8f oacc[4];
    for (int dt = 0; dt < 4; ++dt) for (int j = 0; j < 8; ++j) oacc[dt][j] = 0.f;
#pragma unroll
    for (int ks = 0; ks < 7; ++ks) {
      v16bf pf;
#pragma unroll
      for (int v = 0; v < 8; ++v) {
        int kb = (v & 3) * 2 + (v >> 2) * 16 + half * 8 + ks * 32;
        pf[2 * v]     = P[l16 * LDP + kb];
        pf[2 * v + 1] = P[l16 * LDP + kb + 1];
      }
#pragma unroll
      for (int dt = 0; dt < 4; ++dt) {
        int d = dt * 16 + l16;
        v16bf vf;
#pragma unroll
        for (int v = 0; v < 8; ++v) {
          int key = ks * 32 + (v & 3) * 2 + (v >> 2) * 16 + half * 8;        // P(key>=TT)==0 exactly
          int k0c = (key     < TT) ? key     : TT - 1;
          int k1c = (key + 1 < TT) ? key + 1 : TT - 1;
          vf[2 * v]     = qkv[(size_t)(b * TT + k0c) * row3 + 2 * CC + h * HDIM + d];
          vf[2 * v + 1] = qkv[(size_t)(b * TT + k1c) * row3 + 2 * CC + h * HDIM + d];
        }
        oacc[dt] = __builtin_amdgcn_wmma_f32_16x16x32_bf16(
            false, pf, false, vf, (short)0, oacc[dt], false, false);
      }
    }
#pragma unroll
    for (int dt = 0; dt < 4; ++dt) {
      int d = dt * 16 + l16;
#pragma unroll
      for (int v = 0; v < 8; ++v) {
        int tq = qt * 16 + v + half * 8;
        if (tq < TT)
          o[(size_t)(b * TT + tq) * CC + h * HDIM + d] = f2bf(oacc[dt][v]);
      }
    }
  }
}

// ---------------------------------------------------------------------------
// Reducer pieces
// ---------------------------------------------------------------------------
__global__ void k_sumsq(const float* __restrict__ x, float* __restrict__ ss) {
  int row = blockIdx.x;
  __shared__ float red[256];
  float s = 0.f;
  for (int c = threadIdx.x; c < CC; c += 256) {
    float v = x[(size_t)row * CC + c]; s += v * v;
  }
  red[threadIdx.x] = s; __syncthreads();
  for (int o = 128; o > 0; o >>= 1) {
    if (threadIdx.x < o) red[threadIdx.x] += red[threadIdx.x + o];
    __syncthreads();
  }
  if (threadIdx.x == 0) ss[row] = red[0];
}

// rank-based top-K (stable: value desc, index asc), cls token forced to rank 0
__global__ void k_topk(const float* __restrict__ ss, int* __restrict__ keep) {
  int b = blockIdx.x;
  __shared__ float sb[TT];
  int t = threadIdx.x;
  if (t < TT) sb[t] = (t == 0) ? INFINITY : sqrtf(ss[b * TT + t] + 1e-6f);
  __syncthreads();
  if (t < TT) {
    float bv = sb[t];
    int r = 0;
    for (int j = 0; j < TT; ++j) {
      float o = sb[j];
      r += (o > bv) || (o == bv && j < t);
    }
    if (r < KK) keep[b * KK + r] = t;
  }
}

__global__ void k_assign(const float* __restrict__ x, const float* __restrict__ ss,
                         const int* __restrict__ keep, int* __restrict__ assign) {
  int row = blockIdx.x;                 // b*TT + t
  int b = row / TT;
  __shared__ float xt[CC];
  __shared__ float sc[256];
  __shared__ int   si[256];
  for (int c = threadIdx.x; c < CC; c += 256) xt[c] = x[(size_t)row * CC + c];
  __syncthreads();
  float best = -3.0e38f; int bi = 0x7fffffff;
  int k = threadIdx.x;
  if (k < KK) {
    int kid = keep[b * KK + k];
    const float* cr = x + (size_t)(b * TT + kid) * CC;
    float dot = 0.f;
    for (int c = 0; c < CC; ++c) dot += xt[c] * cr[c];
    best = dot / fmaxf(sqrtf(ss[b * TT + kid]), 1e-12f);
    bi = k;
  }
  sc[threadIdx.x] = best; si[threadIdx.x] = bi; __syncthreads();
  for (int o = 128; o > 0; o >>= 1) {
    if (threadIdx.x < o) {
      float s2 = sc[threadIdx.x + o]; int i2 = si[threadIdx.x + o];
      if (s2 > sc[threadIdx.x] || (s2 == sc[threadIdx.x] && i2 < si[threadIdx.x])) {
        sc[threadIdx.x] = s2; si[threadIdx.x] = i2;
      }
    }
    __syncthreads();
  }
  if (threadIdx.x == 0) assign[row] = si[0];
}

__global__ void k_override(const int* __restrict__ keep, int* __restrict__ assign) {
  int b = blockIdx.x; int k = threadIdx.x;
  if (k < KK) assign[b * TT + keep[b * KK + k]] = k;
}

__global__ void k_merge(const float* __restrict__ x, const int* __restrict__ assign,
                        float* __restrict__ xm) {
  int blk = blockIdx.x;                 // b*KK + k
  int b = blk / KK, k = blk % KK;
  float acc[4] = {0.f, 0.f, 0.f, 0.f};
  int cnt = 0;
  const float* xb = x + (size_t)b * TT * CC;
  const int* ab = assign + b * TT;
  for (int t = 0; t < TT; ++t) {
    if (ab[t] == k) {
      ++cnt;
#pragma unroll
      for (int i = 0; i < 4; ++i)
        acc[i] += xb[(size_t)t * CC + threadIdx.x + i * 256];
    }
  }
  float inv = 1.f / fmaxf((float)cnt, 1.f);
#pragma unroll
  for (int i = 0; i < 4; ++i)
    xm[(size_t)blk * CC + threadIdx.x + i * 256] = acc[i] * inv;
}

// ---------------------------------------------------------------------------
// Host orchestration
// ---------------------------------------------------------------------------
extern "C" void kernel_launch(void* const* d_in, const int* in_sizes, int n_in,
                              void* d_out, int out_size, void* d_ws, size_t ws_size,
                              hipStream_t stream) {
  (void)in_sizes; (void)n_in; (void)out_size; (void)ws_size;
  const float* x      = (const float*)d_in[0];
  const float* w_qkv  = (const float*)d_in[1];
  const float* b_qkv  = (const float*)d_in[2];
  const float* w_proj = (const float*)d_in[3];
  const float* b_proj = (const float*)d_in[4];
  const float* ln1_g  = (const float*)d_in[5];
  const float* ln1_b  = (const float*)d_in[6];
  const float* ln2_g  = (const float*)d_in[7];
  const float* ln2_b  = (const float*)d_in[8];
  const float* w_fc1  = (const float*)d_in[9];
  const float* b_fc1  = (const float*)d_in[10];
  const float* w_fc2  = (const float*)d_in[11];
  const float* b_fc2  = (const float*)d_in[12];
  float* outp = (float*)d_out;

  char* ws = (char*)d_ws;
  size_t off = 0;
  auto alloc = [&](size_t bytes) { size_t o = off; off += (bytes + 255) & ~(size_t)255; return o; };

  // bf16 weight copies (persist for whole call)
  __bf16* wqb = (__bf16*)(ws + alloc((size_t)3 * CC * CC * 2));
  __bf16* wpb = (__bf16*)(ws + alloc((size_t)CC * CC * 2));
  __bf16* w1b = (__bf16*)(ws + alloc((size_t)HIDN * CC * 2));
  __bf16* w2b = (__bf16*)(ws + alloc((size_t)CC * HIDN * 2));
  // big region: qkv (bf16, BT x 3C) then reused for MLP activations (BK x HID)
  size_t big = (size_t)BK * HIDN * 2;           // 82.3 MB >= BT*3C*2
  __bf16* qkv_b = (__bf16*)(ws + alloc(big));
  __bf16* hbuf  = (__bf16*)(ws + alloc((size_t)BT * CC * 2));  // h -> o -> h2
  float*  xnew  = (float*)(ws + alloc((size_t)BT * CC * 4));
  float*  xm    = (float*)(ws + alloc((size_t)BK * CC * 4));
  float*  ssq   = (float*)(ws + alloc((size_t)BT * 4));
  int*    keep  = (int*)(ws + alloc((size_t)BB * KK * 4));
  int*    asg   = (int*)(ws + alloc((size_t)BT * 4));
  __bf16* act   = qkv_b;    // qkv dead after attention
  __bf16* obuf  = hbuf;     // h dead after QKV GEMM
  __bf16* h2    = hbuf;     // o dead after proj GEMM

  dim3 blk(256);

  // 0) weights -> bf16
  k_w2bf<<<(3 * CC * CC) / 1024, blk, 0, stream>>>(w_qkv, wqb, 3 * CC * CC);
  k_w2bf<<<(CC * CC) / 1024, blk, 0, stream>>>(w_proj, wpb, CC * CC);
  k_w2bf<<<(HIDN * CC) / 1024, blk, 0, stream>>>(w_fc1, w1b, HIDN * CC);
  k_w2bf<<<(CC * HIDN) / 1024, blk, 0, stream>>>(w_fc2, w2b, CC * HIDN);

  // 1) h = LN1(x) -> bf16
  k_layernorm_bf16<<<BT, blk, 0, stream>>>(x, ln1_g, ln1_b, hbuf);

  // 2) qkv = h @ w_qkv^T + b_qkv   [BT, 3C] bf16
  k_gemm_wmma<<<dim3(3 * CC / 64, (BT + 127) / 128), blk, 0, stream>>>(
      hbuf, wqb, b_qkv, nullptr, nullptr, qkv_b, BT, 3 * CC, CC, 0);

  // 3) attention -> o bf16 [BT, C]
  k_attention<<<BB * NH, blk, 0, stream>>>(qkv_b, obuf);

  // 4) x_new = x + o @ w_proj^T + b_proj   fp32
  k_gemm_wmma<<<dim3(CC / 64, (BT + 127) / 128), blk, 0, stream>>>(
      obuf, wpb, b_proj, x, xnew, nullptr, BT, CC, CC, 0);

  // 5) reducer
  k_sumsq<<<BT, blk, 0, stream>>>(xnew, ssq);
  k_topk<<<BB, blk, 0, stream>>>(ssq, keep);
  k_assign<<<BT, blk, 0, stream>>>(xnew, ssq, keep, asg);
  k_override<<<BB, blk, 0, stream>>>(keep, asg);
  k_merge<<<BB * KK, blk, 0, stream>>>(xnew, asg, xm);

  // 6) MLP on merged tokens
  k_layernorm_bf16<<<BK, blk, 0, stream>>>(xm, ln2_g, ln2_b, h2);
  k_gemm_wmma<<<dim3(HIDN / 64, (BK + 127) / 128), blk, 0, stream>>>(
      h2, w1b, b_fc1, nullptr, nullptr, act, BK, HIDN, CC, 1);        // GELU
  k_gemm_wmma<<<dim3(CC / 64, (BK + 127) / 128), blk, 0, stream>>>(
      act, w2b, b_fc2, xm, outp, nullptr, BK, CC, HIDN, 0);           // +resid
}
